// NeuralRoughSimulator_16587163697868
// MI455X (gfx1250) — compile-verified, matrix operand order checked by reading, not run on hardware
//
#include <hip/hip_runtime.h>

// ---------------------------------------------------------------------------
// Neural rough-vol simulator for MI455X (gfx1250, wave32, WMMA), v3.
//
// Transposed formulation: D = W (A-matrix, 16 feature-rows x K) * X (B-matrix,
// K x 16 paths). f16 D layout == f16 B layout, so a layer's packed output IS
// the next layer's B operand: no LDS shuffle between layers. Weights/biases
// are loop-invariant -> compiler keeps them register-resident (>256 VGPRs via
// VGPR-MSB), which is optimal at ~1 wave/SIMD (512 waves, grid-limited).
//
// v3: the two independent MLPs (drift, diffusion) are evaluated tile-
// interleaved in one fused routine, giving the scheduler two independent
// wmma -> pk_add -> v_tanh_f32 -> cvt_pk_rtz chains to overlap (fills the
// WMMA->VALU hazard slots; no cross-wave latency hiding exists at 1 wave/SIMD).
// Per step per wave (16 paths): 44x v_wmma_f32_16x16x32_f16.
// ---------------------------------------------------------------------------

typedef __attribute__((ext_vector_type(16))) _Float16 v16h;
typedef __attribute__((ext_vector_type(2)))  _Float16 v2h;
typedef __attribute__((ext_vector_type(8)))  float    v8f;

#define WAVES    2
#define BLOCK    (32 * WAVES)
#define TILE_M   16
#define LD_W0    34   // w0 [64][34], cols 15..33 zero (K padded 15->32)
#define LD_W     66   // w1/w2 [64][66], cols 64..65 zero; odd dword stride
#define LD_W3    66   // w3 tile [16][66]: rows 0 and 8 = w3, rest zero
#define KAPPA    2.72f
#define THETA    (-3.5f)
#define LOG2E    1.4426950408889634f
#define LN2      0.6931471805599453f

__device__ __forceinline__ float fast_tanh(float x) {
#if __has_builtin(__builtin_amdgcn_tanhf)
  return __builtin_amdgcn_tanhf(x);            // v_tanh_f32
#else
  float y;
  asm volatile("v_tanh_f32 %0, %1" : "=v"(y) : "v"(x));
  return y;
#endif
}

__device__ __forceinline__ float fast_sigmoid(float x) {
  return __builtin_amdgcn_rcpf(1.0f + __builtin_amdgcn_exp2f(-x * LOG2E));
}

__device__ __forceinline__ v2h pk2(float a, float b) {
  // v_cvt_pk_rtz_f16_f32; builtin returns __fp16x2 -> bit-cast to _Float16x2
  return __builtin_bit_cast(v2h, __builtin_amdgcn_cvt_pkrtz(a, b));
}

__device__ __forceinline__ v8f wmma_f16(v16h a, v16h b, v8f c) {
  return __builtin_amdgcn_wmma_f32_16x16x32_f16(false, a, false, b, (short)0, c,
                                                false, false);
}

// A-layout 16x32 f16 fragment loader (ISA 7.12.2): lane holds row lane&15;
// lanes 0-15 supply K {0..7,16..23}+k0, lanes 16-31 K {8..15,24..31}+k0.
__device__ __forceinline__ v16h lds_fragA(const _Float16* row0, int lane, int ld, int k0) {
  const _Float16* r = row0 + (lane & 15) * ld + k0 + ((lane >> 4) & 1) * 8;
  union { v16h v; v2h p[8]; } u;
#pragma unroll
  for (int i = 0; i < 4; ++i) {
    u.p[i]     = *(const v2h*)(r + 2 * i);
    u.p[4 + i] = *(const v2h*)(r + 16 + 2 * i);
  }
  return u.v;
}

struct Q { v2h q0, q1, q2, q3; };

// bias-add + tanh + pack rows (2j,2j+1) -> B-operand halves for the next layer
__device__ __forceinline__ Q act_pack(v8f c, const float* __restrict__ b) {
  const float4 blo = *(const float4*)(b);      // loop-invariant -> hoisted to regs
  const float4 bhi = *(const float4*)(b + 4);
  Q r;
  r.q0 = pk2(fast_tanh(c[0] + blo.x), fast_tanh(c[1] + blo.y));
  r.q1 = pk2(fast_tanh(c[2] + blo.z), fast_tanh(c[3] + blo.w));
  r.q2 = pk2(fast_tanh(c[4] + bhi.x), fast_tanh(c[5] + bhi.y));
  r.q3 = pk2(fast_tanh(c[6] + bhi.z), fast_tanh(c[7] + bhi.w));
  return r;
}

// one output tile (16 features), single-K-chunk layer (K=32)
__device__ __forceinline__ Q ltile1(const _Float16* w, int ld, v16h x,
                                    const float* bias, int lane) {
  v8f c = {};
  c = wmma_f16(lds_fragA(w, lane, ld, 0), x, c);
  return act_pack(c, bias);
}

// one output tile, two-K-chunk layer (K=64)
__device__ __forceinline__ Q ltile2(const _Float16* w, int ld, v16h xlo, v16h xhi,
                                    const float* bias, int lane) {
  v8f c = {};
  c = wmma_f16(lds_fragA(w, lane, ld, 0),  xlo, c);
  c = wmma_f16(lds_fragA(w, lane, ld, 32), xhi, c);
  return act_pack(c, bias);
}

union Y { v16h v; v2h p[8]; };

#define PUT_LO(Y_, r) { Y_.p[0] = r.q0; Y_.p[1] = r.q1; Y_.p[2] = r.q2; Y_.p[3] = r.q3; }
#define PUT_HI(Y_, r) { Y_.p[4] = r.q0; Y_.p[5] = r.q1; Y_.p[6] = r.q2; Y_.p[7] = r.q3; }

// Fused dual-MLP (drift + diffusion), tile-interleaved so the two independent
// dependency chains overlap. Returns both scalar outputs (mirror-preserving:
// value for path (lane&15) lands in every lane).
__device__ __forceinline__ void mlp_dual(v16h x,
    const _Float16* __restrict__ dw0, const _Float16* __restrict__ dw1,
    const _Float16* __restrict__ dw2, const _Float16* __restrict__ dw3,
    const _Float16* __restrict__ fw0, const _Float16* __restrict__ fw1,
    const _Float16* __restrict__ fw2, const _Float16* __restrict__ fw3,
    const float* __restrict__ db0, const float* __restrict__ db1,
    const float* __restrict__ db2,
    const float* __restrict__ fb0, const float* __restrict__ fb1,
    const float* __restrict__ fb2,
    float bd3, float bf3, int lane, int hoff, float& draw, float& fraw) {
  Y dy0l, dy0h, dy1l, dy1h;
  Y fy0l, fy0h, fy1l, fy1h;
  // ---- layer 0 (K=32, 15 valid), drift/diff alternating
  { Q r = ltile1(dw0 + 0 * 16 * LD_W0, LD_W0, x, db0 + 0  + hoff, lane); PUT_LO(dy0l, r); }
  { Q r = ltile1(fw0 + 0 * 16 * LD_W0, LD_W0, x, fb0 + 0  + hoff, lane); PUT_LO(fy0l, r); }
  { Q r = ltile1(dw0 + 1 * 16 * LD_W0, LD_W0, x, db0 + 16 + hoff, lane); PUT_HI(dy0l, r); }
  { Q r = ltile1(fw0 + 1 * 16 * LD_W0, LD_W0, x, fb0 + 16 + hoff, lane); PUT_HI(fy0l, r); }
  { Q r = ltile1(dw0 + 2 * 16 * LD_W0, LD_W0, x, db0 + 32 + hoff, lane); PUT_LO(dy0h, r); }
  { Q r = ltile1(fw0 + 2 * 16 * LD_W0, LD_W0, x, fb0 + 32 + hoff, lane); PUT_LO(fy0h, r); }
  { Q r = ltile1(dw0 + 3 * 16 * LD_W0, LD_W0, x, db0 + 48 + hoff, lane); PUT_HI(dy0h, r); }
  { Q r = ltile1(fw0 + 3 * 16 * LD_W0, LD_W0, x, fb0 + 48 + hoff, lane); PUT_HI(fy0h, r); }
  // ---- layer 1 (K=64)
  { Q r = ltile2(dw1 + 0 * 16 * LD_W, LD_W, dy0l.v, dy0h.v, db1 + 0  + hoff, lane); PUT_LO(dy1l, r); }
  { Q r = ltile2(fw1 + 0 * 16 * LD_W, LD_W, fy0l.v, fy0h.v, fb1 + 0  + hoff, lane); PUT_LO(fy1l, r); }
  { Q r = ltile2(dw1 + 1 * 16 * LD_W, LD_W, dy0l.v, dy0h.v, db1 + 16 + hoff, lane); PUT_HI(dy1l, r); }
  { Q r = ltile2(fw1 + 1 * 16 * LD_W, LD_W, fy0l.v, fy0h.v, fb1 + 16 + hoff, lane); PUT_HI(fy1l, r); }
  { Q r = ltile2(dw1 + 2 * 16 * LD_W, LD_W, dy0l.v, dy0h.v, db1 + 32 + hoff, lane); PUT_LO(dy1h, r); }
  { Q r = ltile2(fw1 + 2 * 16 * LD_W, LD_W, fy0l.v, fy0h.v, fb1 + 32 + hoff, lane); PUT_LO(fy1h, r); }
  { Q r = ltile2(dw1 + 3 * 16 * LD_W, LD_W, dy0l.v, dy0h.v, db1 + 48 + hoff, lane); PUT_HI(dy1h, r); }
  { Q r = ltile2(fw1 + 3 * 16 * LD_W, LD_W, fy0l.v, fy0h.v, fb1 + 48 + hoff, lane); PUT_HI(fy1h, r); }
  // ---- layer 2 (K=64), results back into y0 registers
  { Q r = ltile2(dw2 + 0 * 16 * LD_W, LD_W, dy1l.v, dy1h.v, db2 + 0  + hoff, lane); PUT_LO(dy0l, r); }
  { Q r = ltile2(fw2 + 0 * 16 * LD_W, LD_W, fy1l.v, fy1h.v, fb2 + 0  + hoff, lane); PUT_LO(fy0l, r); }
  { Q r = ltile2(dw2 + 1 * 16 * LD_W, LD_W, dy1l.v, dy1h.v, db2 + 16 + hoff, lane); PUT_HI(dy0l, r); }
  { Q r = ltile2(fw2 + 1 * 16 * LD_W, LD_W, fy1l.v, fy1h.v, fb2 + 16 + hoff, lane); PUT_HI(fy0l, r); }
  { Q r = ltile2(dw2 + 2 * 16 * LD_W, LD_W, dy1l.v, dy1h.v, db2 + 32 + hoff, lane); PUT_LO(dy0h, r); }
  { Q r = ltile2(fw2 + 2 * 16 * LD_W, LD_W, fy1l.v, fy1h.v, fb2 + 32 + hoff, lane); PUT_LO(fy0h, r); }
  { Q r = ltile2(dw2 + 3 * 16 * LD_W, LD_W, dy1l.v, dy1h.v, db2 + 48 + hoff, lane); PUT_HI(dy0h, r); }
  { Q r = ltile2(fw2 + 3 * 16 * LD_W, LD_W, fy1l.v, fy1h.v, fb2 + 48 + hoff, lane); PUT_HI(fy0h, r); }
  // ---- layer 3 (64 -> 1); w3 in rows 0 and 8 -> result in c[0] for both halves
  v8f dc = {};
  v8f fc = {};
  dc = wmma_f16(lds_fragA(dw3, lane, LD_W3, 0),  dy0l.v, dc);
  fc = wmma_f16(lds_fragA(fw3, lane, LD_W3, 0),  fy0l.v, fc);
  dc = wmma_f16(lds_fragA(dw3, lane, LD_W3, 32), dy0h.v, dc);
  fc = wmma_f16(lds_fragA(fw3, lane, LD_W3, 32), fy0h.v, fc);
  draw = dc[0] + bd3;
  fraw = fc[0] + bf3;
}

__global__ __launch_bounds__(BLOCK) void nrs_kernel(
    const float* __restrict__ init_var, const float* __restrict__ dtp,
    const float* __restrict__ dW,
    const float* __restrict__ d_w0, const float* __restrict__ d_b0,
    const float* __restrict__ d_w1, const float* __restrict__ d_b1,
    const float* __restrict__ d_w2, const float* __restrict__ d_b2,
    const float* __restrict__ d_w3, const float* __restrict__ d_b3,
    const float* __restrict__ f_w0, const float* __restrict__ f_b0,
    const float* __restrict__ f_w1, const float* __restrict__ f_b1,
    const float* __restrict__ f_w2, const float* __restrict__ f_b2,
    const float* __restrict__ f_w3, const float* __restrict__ f_b3,
    float* __restrict__ out, int T) {
  // ---- LDS (~48 KB/block, block-shared, read-only after staging)
  __shared__ _Float16 sW0[2][64 * LD_W0];
  __shared__ _Float16 sW1[2][64 * LD_W];
  __shared__ _Float16 sW2[2][64 * LD_W];
  __shared__ _Float16 sW3[2][16 * LD_W3];
  __shared__ float    sB[2][3][64];

  const int tid = threadIdx.x;

  for (int i = tid; i < 64 * LD_W0; i += BLOCK) {
    const int o = i / LD_W0, k = i % LD_W0;
    sW0[0][i] = (k < 15) ? (_Float16)d_w0[o * 15 + k] : (_Float16)0.0f;
    sW0[1][i] = (k < 15) ? (_Float16)f_w0[o * 15 + k] : (_Float16)0.0f;
  }
  for (int i = tid; i < 64 * LD_W; i += BLOCK) {
    const int o = i / LD_W, k = i % LD_W;
    const bool ok = (k < 64);
    sW1[0][i] = ok ? (_Float16)d_w1[o * 64 + k] : (_Float16)0.0f;
    sW1[1][i] = ok ? (_Float16)f_w1[o * 64 + k] : (_Float16)0.0f;
    sW2[0][i] = ok ? (_Float16)d_w2[o * 64 + k] : (_Float16)0.0f;
    sW2[1][i] = ok ? (_Float16)f_w2[o * 64 + k] : (_Float16)0.0f;
  }
  for (int i = tid; i < 16 * LD_W3; i += BLOCK) {
    const int r = i / LD_W3, k = i % LD_W3;
    const bool on = (k < 64) && ((r == 0) || (r == 8));
    sW3[0][i] = on ? (_Float16)d_w3[k] : (_Float16)0.0f;
    sW3[1][i] = on ? (_Float16)f_w3[k] : (_Float16)0.0f;
  }
  for (int i = tid; i < 64; i += BLOCK) {
    sB[0][0][i] = d_b0[i]; sB[0][1][i] = d_b1[i]; sB[0][2][i] = d_b2[i];
    sB[1][0][i] = f_b0[i]; sB[1][1][i] = f_b1[i]; sB[1][2][i] = f_b2[i];
  }
  __syncthreads();

  const int wave = tid >> 5;
  const int lane = tid & 31;
  const int nloc = lane & 15;
  const int hoff = ((lane >> 4) & 1) * 8;      // bias row offset for this half
  const int path = blockIdx.x * (WAVES * TILE_M) + wave * TILE_M + nloc;
  const float dt  = dtp[0];
  const float bd3 = d_b3[0], bf3 = f_b3[0];

  // ---- per-path f32 state; lanes 16-31 mirror lanes 0-15 exactly
  float lv = __builtin_amdgcn_logf(fminf(fmaxf(init_var[path], 0.01f), 1.5f)) * LN2;
  float s1a = 0.f, s1b = 0.f;
  float s2_0 = 0.f, s2_1 = 0.f, s2_2 = 0.f, s2_3 = 0.f;
  float s3_0 = 0.f, s3_1 = 0.f, s3_2 = 0.f, s3_3 = 0.f;
  float s3_4 = 0.f, s3_5 = 0.f, s3_6 = 0.f, s3_7 = 0.f;

  const bool hi = (lane & 16) != 0;
  const v2h zz = {(_Float16)0.0f, (_Float16)0.0f};
  const size_t row = (size_t)path * (size_t)T;

  for (int t = 0; t < T; ++t) {
    // ---- input B-fragment in registers: K = [s1(2), s2(4), s3(8), lv+4, 0..]
    // lanes 0-15 carry K0..7 (and zero K16..23); lanes 16-31 carry K8..15
    Y x;
    x.p[0] = hi ? pk2(s3_2, s3_3)      : pk2(s1a,  s1b);
    x.p[1] = hi ? pk2(s3_4, s3_5)      : pk2(s2_0, s2_1);
    x.p[2] = hi ? pk2(s3_6, s3_7)      : pk2(s2_2, s2_3);
    x.p[3] = hi ? pk2(lv + 4.0f, 0.0f) : pk2(s3_0, s3_1);
    x.p[4] = zz; x.p[5] = zz; x.p[6] = zz; x.p[7] = zz;

    float draw, fraw;
    mlp_dual(x.v,
             sW0[0], sW1[0], sW2[0], sW3[0],
             sW0[1], sW1[1], sW2[1], sW3[1],
             sB[0][0], sB[0][1], sB[0][2],
             sB[1][0], sB[1][1], sB[1][2],
             bd3, bf3, lane, hoff, draw, fraw);

    const float drift_nn  = 0.5f * fast_tanh(draw);
    const float diffusion = 1.5f * fast_sigmoid(fraw) + 0.1f;
    const float dwt = dW[row + t];

    float lvn = lv + KAPPA * (THETA - lv) * dt + drift_nn * dt + diffusion * dwt;
    lvn = fminf(fmaxf(lvn, -5.0f), 1.0f);
    const float d = lvn - lv;

    // ---- signature update (uses OLD s1/s2)
    const float xx00 = dt * dt, xx01 = dt * d, xx11 = d * d;
    const float os1a = s1a, os1b = s1b;
    const float os20 = s2_0, os21 = s2_1, os22 = s2_2, os23 = s2_3;
    const float c6 = 1.0f / 6.0f;
    s3_0 += os20 * dt + 0.5f * os1a * xx00 + dt * xx00 * c6;
    s3_1 += os20 * d  + 0.5f * os1a * xx01 + dt * xx01 * c6;
    s3_2 += os21 * dt + 0.5f * os1a * xx01 + dt * xx01 * c6;
    s3_3 += os21 * d  + 0.5f * os1a * xx11 + dt * xx11 * c6;
    s3_4 += os22 * dt + 0.5f * os1b * xx00 + d * xx00 * c6;
    s3_5 += os22 * d  + 0.5f * os1b * xx01 + d * xx01 * c6;
    s3_6 += os23 * dt + 0.5f * os1b * xx01 + d * xx01 * c6;
    s3_7 += os23 * d  + 0.5f * os1b * xx11 + d * xx11 * c6;
    s2_0 += os1a * dt + 0.5f * xx00;
    s2_1 += os1a * d  + 0.5f * xx01;
    s2_2 += os1b * dt + 0.5f * xx01;
    s2_3 += os1b * d  + 0.5f * xx11;
    s1a += dt; s1b += d;
    lv = lvn;

    if (lane < 16) out[row + t] = __builtin_amdgcn_exp2f(lvn * LOG2E);
  }
}

extern "C" void kernel_launch(void* const* d_in, const int* in_sizes, int n_in,
                              void* d_out, int out_size, void* d_ws, size_t ws_size,
                              hipStream_t stream) {
  const int B = in_sizes[0];
  const int T = in_sizes[2] / B;
  dim3 grid(B / (WAVES * TILE_M));   // 8192 / 32 = 256 blocks, 512 waves
  nrs_kernel<<<grid, BLOCK, 0, stream>>>(
      (const float*)d_in[0],  (const float*)d_in[1],  (const float*)d_in[2],
      (const float*)d_in[3],  (const float*)d_in[4],  (const float*)d_in[5],
      (const float*)d_in[6],  (const float*)d_in[7],  (const float*)d_in[8],
      (const float*)d_in[9],  (const float*)d_in[10], (const float*)d_in[11],
      (const float*)d_in[12], (const float*)d_in[13], (const float*)d_in[14],
      (const float*)d_in[15], (const float*)d_in[16], (const float*)d_in[17],
      (const float*)d_in[18], (float*)d_out, T);
}
